// TransformerNvidiaPolicy_32006096290242
// MI455X (gfx1250) — matrix operand order repaired and checked by reading, not compiled, not run on hardware
//
#include <hip/hip_runtime.h>
#include <hip/hip_bf16.h>

// ---------------------------------------------------------------------------
// Transformer-XL block for MI455X (gfx1250): bf16 WMMA GEMMs + fused
// flash-attention. Tile staging to LDS uses the Tensor Data Mover
// (tensor_load_to_lds / s_wait_tensorcnt) where no transpose is needed.
// ---------------------------------------------------------------------------

typedef __bf16 bf16_t;
typedef __attribute__((ext_vector_type(16))) __bf16 v16bf;
typedef __attribute__((ext_vector_type(8)))  float  v8f;
typedef __attribute__((ext_vector_type(4)))  unsigned int u32x4;
typedef __attribute__((ext_vector_type(8)))  int i32x8;
typedef __attribute__((ext_vector_type(4)))  int i32x4;

constexpr int Lc  = 2;
constexpr int Tc  = 1024;
constexpr int Bc  = 4;
constexpr int Dc  = 1024;
constexpr int Hc  = 16;
constexpr int DHc = 64;
constexpr int DIc = 4096;
constexpr int Mc  = 1024;
constexpr int KLc = Mc + Tc;        // 2048
constexpr int HDHc = Hc * DHc;      // 1024
constexpr float SCALEc = 0.125f;    // 1/sqrt(64)

union V16u { v16bf v; uint4 q[2]; };

__device__ __forceinline__ v8f v8f_zero() {
    v8f z = {0.f, 0.f, 0.f, 0.f, 0.f, 0.f, 0.f, 0.f};
    return z;
}

__device__ __forceinline__ v8f wmma_bf16(v16bf a, v16bf b, v8f c) {
    return __builtin_amdgcn_wmma_f32_16x16x32_bf16(
        /*neg_a=*/false, a, /*neg_b=*/false, b,
        /*c_mod=*/(short)0, c, /*reuse_a=*/false, /*reuse_b=*/false);
}

// A fragment (16x32 bf16): lane L -> row M = mbase + (L&15); K halves:
// k = (e&8 ? 16:0) + ((L&16)?8:0) + (e&7)  -> two contiguous 8-half runs.
__device__ __forceinline__ v16bf load_frag_a(const bf16_t* sm, int mbase,
                                             int ldk, int kofs) {
    int L = threadIdx.x & 31;
    const bf16_t* p = sm + (mbase + (L & 15)) * ldk + kofs + ((L & 16) >> 1);
    V16u r;
    r.q[0] = *(const uint4*)(p);
    r.q[1] = *(const uint4*)(p + 16);
    return r.v;
}

// B fragment (32x16 bf16) from LDS stored [n][k] (k contiguous):
// lane L -> col N = nbase + (L&15); k = (e&8 ? 16:0) + ((L&16)?8:0) + (e&7).
__device__ __forceinline__ v16bf load_frag_b(const bf16_t* sm, int nbase,
                                             int ldk, int kofs) {
    int L = threadIdx.x & 31;
    const bf16_t* p = sm + (nbase + (L & 15)) * ldk + kofs + ((L & 16) >> 1);
    V16u r;
    r.q[0] = *(const uint4*)(p);
    r.q[1] = *(const uint4*)(p + 16);
    return r.v;
}

// ---------------------------------------------------------------------------
// Tensor Data Mover: load a 2D tile (tile_w 8-byte units x tile_h rows,
// row stride stride_u 8-byte units) from global into LDS at lds_addr.
// D# packing per CDNA5 ISA: group0 = {flags, lds_addr, global_addr, type=2},
// group1 = {data_size=8B, tensor dims, tile dims, strides}. Groups 2/3 zero
// (<=2D tensor). Issue from one wave; fence with s_wait_tensorcnt.
// ---------------------------------------------------------------------------
__device__ __forceinline__ void tdm_load_tile(unsigned lds_addr,
                                              const void* gptr,
                                              unsigned tile_w_u,
                                              unsigned tile_h,
                                              unsigned stride_u) {
    unsigned long long ga = (unsigned long long)(uintptr_t)gptr;
    u32x4 g0;
    g0[0] = 1u;                                      // count=1 (valid), user mode
    g0[1] = lds_addr;                                // lds_addr [63:32]
    g0[2] = (unsigned)(ga & 0xffffffffull);          // global_addr [95:64]
    g0[3] = (unsigned)((ga >> 32) & 0x1ffffffull)    // global_addr [120:96]
            | (2u << 30);                            // type = 2 ("image")
    const unsigned td0 = tile_w_u;                   // tensor_dim0 (OOB bound)
    const unsigned td1 = 0x100000u;                  // tensor_dim1: generous
    i32x8 g1;
    g1[0] = (int)(3u << 16);                         // data_size = 3 (8 bytes)
    g1[1] = (int)((td0 & 0xffffu) << 16);            // tensor_dim0[15:0]
    g1[2] = (int)((td0 >> 16) | ((td1 & 0xffffu) << 16));
    g1[3] = (int)((td1 >> 16) | ((tile_w_u & 0xffffu) << 16)); // tile_dim0
    g1[4] = (int)(tile_h & 0xffffu);                 // tile_dim1 (tile_dim2=0)
    g1[5] = (int)stride_u;                           // tensor_dim0_stride[31:0]
    g1[6] = 0;                                       // stride0 hi / stride1 lo
    g1[7] = 0;
    i32x4 z4 = {0, 0, 0, 0};
#if defined(__clang_major__) && (__clang_major__ >= 23)
    i32x8 z8 = {0, 0, 0, 0, 0, 0, 0, 0};
    __builtin_amdgcn_tensor_load_to_lds(g0, g1, z4, z4, z8, 0);
#else
    __builtin_amdgcn_tensor_load_to_lds(g0, g1, z4, z4, 0);
#endif
}

__device__ __forceinline__ unsigned lds_addr_of(const void* p) {
    // Generic LDS pointer: low 32 bits are the LDS byte address.
    return (unsigned)(uintptr_t)p;
}

// ---------------------------------------------------------------------------
// Generic bf16 GEMM: C[M x N] = A[M x K] * B[K x N]  (+bias, relu)
// Block tile 256(M) x 64(N), K-step 32, 256 threads = 8 waves.
// Wave w owns rows [32w, 32w+32): 2 A-frags x 4 B-frags -> 8 WMMA / k-step.
// A tile staged by TDM; B tile loaded+transposed by ALU (TDM can't transpose).
// Requires M%256==0, N%64==0, K%32==0 (true for all shapes here).
// ---------------------------------------------------------------------------
#define TBM 256
#define TBN 64
#define TBK 32

__global__ __launch_bounds__(256)
void gemm_bf16_kernel(const bf16_t* __restrict__ A, int lda,
                      const bf16_t* __restrict__ B, int ldb,
                      float* __restrict__ Cf, bf16_t* __restrict__ Cb, int ldc,
                      const float* __restrict__ bias, int relu, int Kdim) {
    __shared__ bf16_t sA[TBM * TBK];   // [m][k]   (16 KB, TDM-filled)
    __shared__ bf16_t sB[TBN * TBK];   // [n][k]   (transposed on load)

    const int t  = threadIdx.x;
    const int w  = t >> 5;
    const int m0 = blockIdx.y * TBM;
    const int n0 = blockIdx.x * TBN;

    v8f acc[2][4];
#pragma unroll
    for (int s = 0; s < 2; s++)
#pragma unroll
        for (int i = 0; i < 4; i++) acc[s][i] = v8f_zero();

    const unsigned sA_lds = lds_addr_of(sA);

    for (int k0 = 0; k0 < Kdim; k0 += TBK) {
        // A tile via Tensor Data Mover: 256 rows x 64B, row stride lda*2 B.
        if (w == 0) {
            tdm_load_tile(sA_lds, &A[(size_t)m0 * lda + k0],
                          /*tile_w(8B)=*/TBK * 2 / 8, /*tile_h=*/TBM,
                          /*stride(8B)=*/(unsigned)(lda >> 2));
        }
        // B tile: 32x64, transpose into [n][k]
        {
            int kk = t >> 3;          // 0..31
            int nn = (t & 7) << 3;    // 0..56
            uint4 q = *(const uint4*)(&B[(size_t)(k0 + kk) * ldb + n0 + nn]);
            const bf16_t* h = (const bf16_t*)&q;
#pragma unroll
            for (int j = 0; j < 8; j++) sB[(nn + j) * TBK + kk] = h[j];
        }
        if (k0 + TBK < Kdim)
            __builtin_prefetch(&B[(size_t)(k0 + TBK + (t >> 3)) * ldb + n0], 0, 1);
        if (w == 0) __builtin_amdgcn_s_wait_tensorcnt(0);
        __syncthreads();

        v16bf a0 = load_frag_a(sA, w * 32, TBK, 0);
        v16bf a1 = load_frag_a(sA, w * 32 + 16, TBK, 0);
#pragma unroll
        for (int nt = 0; nt < 4; nt++) {
            v16bf b = load_frag_b(sB, nt * 16, TBK, 0);
            acc[0][nt] = wmma_bf16(a0, b, acc[0][nt]);
            acc[1][nt] = wmma_bf16(a1, b, acc[1][nt]);
        }
        __syncthreads();
    }

    // Epilogue: C layout per ISA: VGPR r -> M=r (lanes 0-15) / r+8 (16-31)
    const int L      = t & 31;
    const int lane_n = L & 15;
    const int hi8    = (L & 16) ? 8 : 0;
#pragma unroll
    for (int s = 0; s < 2; s++) {
#pragma unroll
        for (int nt = 0; nt < 4; nt++) {
            int col = n0 + nt * 16 + lane_n;
            float bv = bias ? bias[col] : 0.f;
#pragma unroll
            for (int r = 0; r < 8; r++) {
                int row = m0 + w * 32 + s * 16 + r + hi8;
                float v = acc[s][nt][r] + bv;
                if (relu) v = fmaxf(v, 0.f);
                if (Cf) Cf[(size_t)row * ldc + col] = v;
                else    Cb[(size_t)row * ldc + col] = (bf16_t)v;
            }
        }
    }
}

// ---------------------------------------------------------------------------
// Fused flash attention per (b, h, 64-row query tile).
// score = scale * (Q_w . K^T  +  Q_r[src] . RK[src]^T), causal mask j<=i+M,
// online softmax, O = P.V.  The reference's view-based rel_shift is a pure
// (b,h)->(bs,hs) slice permutation: p = b*H + h + B; bs = p/(H+1);
// ns = p%(H+1); src = (bs, ns-1), zero contribution when ns==0.
// Block: 128 threads = 4 waves; wave w owns rows [16w,16w+16).
// Q/K/RK tiles staged by TDM (wave 0); V staged with an ALU transpose.
// ---------------------------------------------------------------------------
__global__ __launch_bounds__(128)
void attn_kernel(const bf16_t* __restrict__ Qw, const bf16_t* __restrict__ Qr,
                 const bf16_t* __restrict__ Kb, const bf16_t* __restrict__ Vb,
                 const bf16_t* __restrict__ rkb, bf16_t* __restrict__ av) {
    __shared__ bf16_t sQ[64 * 64];
    __shared__ bf16_t sQr[64 * 64];
    __shared__ bf16_t sK[64 * 64];    // [j][d]
    __shared__ bf16_t sRK[64 * 64];   // [j][d]
    __shared__ bf16_t sVt[64 * 64];   // [d][j] (transposed)
    __shared__ bf16_t sP[4 * 16 * 64];// per-wave P tiles [m][j]

    const int t = threadIdx.x, w = t >> 5, L = t & 31;
    const int it = blockIdx.x, h = blockIdx.y, b = blockIdx.z;
    const int i0 = it * 64;

    const int p  = b * Hc + h + Bc;
    const int bs = p / (Hc + 1);
    const int ns = p % (Hc + 1);
    const bool use_src = (ns != 0);
    const int hs = use_src ? (ns - 1) : 0;

    // Stage Q tiles via TDM (both tiles are 64 contiguous rows -> 8KB 1D copy)
    const bf16_t* Qbase  = Qw + ((size_t)(b * Hc + h) * Tc + i0) * DHc;
    const bf16_t* Qrbase = Qr + ((size_t)(bs * Hc + hs) * Tc + i0) * DHc;
    if (w == 0) {
        tdm_load_tile(lds_addr_of(sQ), Qbase, 1024, 1, 1024);
        if (use_src) tdm_load_tile(lds_addr_of(sQr), Qrbase, 1024, 1, 1024);
        __builtin_amdgcn_s_wait_tensorcnt(0);
    }
    __syncthreads();

    v16bf aQ0 = load_frag_a(sQ, w * 16, 64, 0);
    v16bf aQ1 = load_frag_a(sQ, w * 16, 64, 32);
    v16bf aR0 = aQ0, aR1 = aQ1;
    if (use_src) {
        aR0 = load_frag_a(sQr, w * 16, 64, 0);
        aR1 = load_frag_a(sQr, w * 16, 64, 32);
    }

    float mrow[8], lrow[8];
    v8f o[4];
#pragma unroll
    for (int r = 0; r < 8; r++) { mrow[r] = -INFINITY; lrow[r] = 0.f; }
#pragma unroll
    for (int nt = 0; nt < 4; nt++) o[nt] = v8f_zero();

    const int lane_n = L & 15;
    const int hi8    = (L & 16) ? 8 : 0;

    int jt_end = (i0 + 63 + Mc) / 64;
    if (jt_end > KLc / 64 - 1) jt_end = KLc / 64 - 1;

    for (int jt = 0; jt <= jt_end; jt++) {
        const int j0 = jt * 64;
        const bf16_t* Kbase = Kb + ((size_t)(b * Hc + h) * KLc + j0) * DHc;
        const bf16_t* Vbase = Vb + ((size_t)(b * Hc + h) * KLc + j0) * DHc;
        // K tile (contiguous 8KB) and RK tile (2D: 128B rows, 2KB stride) via TDM
        if (w == 0) {
            tdm_load_tile(lds_addr_of(sK), Kbase, 1024, 1, 1024);
            if (use_src)
                tdm_load_tile(lds_addr_of(sRK),
                              &rkb[(size_t)j0 * HDHc + hs * DHc],
                              /*w=*/DHc * 2 / 8, /*h=*/64,
                              /*stride=*/HDHc * 2 / 8);
        }
        // V tile with transpose (ALU path; TDM cannot transpose)
        for (int c = t; c < 512; c += 128) {
            int row = c >> 3, dd = (c & 7) << 3;
            uint4 vq = *(const uint4*)&Vbase[row * DHc + dd];
            const bf16_t* hh = (const bf16_t*)&vq;
#pragma unroll
            for (int jj = 0; jj < 8; jj++) sVt[(dd + jj) * 64 + row] = hh[jj];
        }
        if (w == 0) __builtin_amdgcn_s_wait_tensorcnt(0);
        __syncthreads();

        // S tile (16x64 per wave) via WMMA; BD term chained into same accum
        float sreg[4][8];
#pragma unroll
        for (int nt = 0; nt < 4; nt++) {
            v8f s = v8f_zero();
            s = wmma_bf16(aQ0, load_frag_b(sK, nt * 16, 64, 0), s);
            s = wmma_bf16(aQ1, load_frag_b(sK, nt * 16, 64, 32), s);
            if (use_src) {
                s = wmma_bf16(aR0, load_frag_b(sRK, nt * 16, 64, 0), s);
                s = wmma_bf16(aR1, load_frag_b(sRK, nt * 16, 64, 32), s);
            }
#pragma unroll
            for (int r = 0; r < 8; r++) sreg[nt][r] = s[r];
        }

        // scale + causal mask (mask true iff j >= i + 1 + M)
#pragma unroll
        for (int nt = 0; nt < 4; nt++) {
            int jg = j0 + nt * 16 + lane_n;
#pragma unroll
            for (int r = 0; r < 8; r++) {
                int ig = i0 + w * 16 + r + hi8;
                float v = sreg[nt][r] * SCALEc;
                sreg[nt][r] = (jg > ig + Mc) ? -1.0e30f : v;
            }
        }

        // online softmax (row spread over a 16-lane half-group)
#pragma unroll
        for (int r = 0; r < 8; r++) {
            float mx = fmaxf(fmaxf(sreg[0][r], sreg[1][r]),
                             fmaxf(sreg[2][r], sreg[3][r]));
#pragma unroll
            for (int off = 1; off < 16; off <<= 1)
                mx = fmaxf(mx, __shfl_xor(mx, off, 32));
            float mnew = fmaxf(mrow[r], mx);
            float corr = expf(mrow[r] - mnew);
            float lsum = 0.f;
#pragma unroll
            for (int nt = 0; nt < 4; nt++) {
                float pv = expf(sreg[nt][r] - mnew);
                sreg[nt][r] = pv;
                lsum += pv;
            }
#pragma unroll
            for (int off = 1; off < 16; off <<= 1)
                lsum += __shfl_xor(lsum, off, 32);
            lrow[r] = lrow[r] * corr + lsum;
            mrow[r] = mnew;
#pragma unroll
            for (int nt = 0; nt < 4; nt++) o[nt][r] *= corr;
        }

        // P -> per-wave LDS region, reload as A-frag, O += P.V
        bf16_t* myP = sP + w * 16 * 64;
#pragma unroll
        for (int nt = 0; nt < 4; nt++)
#pragma unroll
            for (int r = 0; r < 8; r++)
                myP[(r + hi8) * 64 + nt * 16 + lane_n] = (bf16_t)sreg[nt][r];

        v16bf aP0 = load_frag_a(myP, 0, 64, 0);
        v16bf aP1 = load_frag_a(myP, 0, 64, 32);
#pragma unroll
        for (int nt = 0; nt < 4; nt++) {
            o[nt] = wmma_bf16(aP0, load_frag_b(sVt, nt * 16, 64, 0), o[nt]);
            o[nt] = wmma_bf16(aP1, load_frag_b(sVt, nt * 16, 64, 32), o[nt]);
        }
        __syncthreads();
    }

    // normalize and emit av[T, B, H*DH] in bf16 (feeds output-proj GEMM)
#pragma unroll
    for (int r = 0; r < 8; r++) {
        float inv = 1.f / lrow[r];
        int ig = i0 + w * 16 + r + hi8;
#pragma unroll
        for (int nt = 0; nt < 4; nt++) {
            int d = nt * 16 + lane_n;
            av[((size_t)ig * Bc + b) * HDHc + h * DHc + d] =
                (bf16_t)(o[nt][r] * inv);
        }
    }
}

// ---------------------------------------------------------------------------
// Elementwise helpers
// ---------------------------------------------------------------------------
__global__ void cast_f2b_kernel(const float* __restrict__ in,
                                bf16_t* __restrict__ out, size_t n) {
    size_t i = (size_t)blockIdx.x * 256 + threadIdx.x;
    if (i < n) out[i] = (bf16_t)in[i];
}

__global__ void pos_emb_kernel(bf16_t* __restrict__ pe) {
    size_t i = (size_t)blockIdx.x * 256 + threadIdx.x;
    if (i >= (size_t)KLc * (Dc / 2)) return;
    int d2 = (int)(i % (Dc / 2));
    int k  = (int)(i / (Dc / 2));
    float invf = expf(-(2.0f * d2 / (float)Dc) * 9.210340371976184f);
    float s = (float)(KLc - 1 - k) * invf;
    pe[(size_t)k * Dc + d2]          = (bf16_t)sinf(s);
    pe[(size_t)k * Dc + Dc / 2 + d2] = (bf16_t)cosf(s);
}

// split wh = [K,B,3*H*DH] into K/V (all k) and biased Q_w/Q_r (k >= M)
__global__ void split_qkv_kernel(const float* __restrict__ wh,
                                 const float* __restrict__ rwbias,
                                 const float* __restrict__ rrbias,
                                 bf16_t* __restrict__ Qw, bf16_t* __restrict__ Qr,
                                 bf16_t* __restrict__ Kb, bf16_t* __restrict__ Vb) {
    size_t idx = (size_t)blockIdx.x * 256 + threadIdx.x;
    if (idx >= (size_t)KLc * Bc * HDHc) return;
    int d = (int)(idx % DHc);
    int h = (int)((idx / DHc) % Hc);
    int b = (int)((idx / HDHc) % Bc);
    int k = (int)(idx / ((size_t)Bc * HDHc));
    const float* row = wh + ((size_t)k * Bc + b) * (3 * HDHc);
    size_t bh = (size_t)(b * Hc + h);
    Kb[(bh * KLc + k) * DHc + d] = (bf16_t)row[HDHc + h * DHc + d];
    Vb[(bh * KLc + k) * DHc + d] = (bf16_t)row[2 * HDHc + h * DHc + d];
    if (k >= Mc) {
        int i = k - Mc;
        float qv = row[h * DHc + d];
        Qw[(bh * Tc + i) * DHc + d] = (bf16_t)(qv + rwbias[h * DHc + d]);
        Qr[(bh * Tc + i) * DHc + d] = (bf16_t)(qv + rrbias[h * DHc + d]);
    }
}

// LayerNorm over rows of length D: out = LN(x + y)*g + b; optional bf16 copy
// and optional second fp32 destination (hids -> new_mems region of d_out).
__global__ __launch_bounds__(256)
void ln_kernel(const float* __restrict__ x, const float* __restrict__ y,
               const float* __restrict__ g, const float* __restrict__ bta,
               float* __restrict__ out, bf16_t* __restrict__ outb,
               float* __restrict__ out2) {
    __shared__ float red[256];
    const int row = blockIdx.x;
    const float* xr = x + (size_t)row * Dc;
    const float* yr = y ? y + (size_t)row * Dc : nullptr;
    float v[4];
    float s = 0.f;
#pragma unroll
    for (int i = 0; i < 4; i++) {
        int c = threadIdx.x + i * 256;
        v[i] = xr[c] + (yr ? yr[c] : 0.f);
        s += v[i];
    }
    red[threadIdx.x] = s;
    __syncthreads();
    for (int st = 128; st > 0; st >>= 1) {
        if (threadIdx.x < st) red[threadIdx.x] += red[threadIdx.x + st];
        __syncthreads();
    }
    float mu = red[0] / Dc;
    __syncthreads();
    float vs = 0.f;
#pragma unroll
    for (int i = 0; i < 4; i++) { float dd = v[i] - mu; vs += dd * dd; }
    red[threadIdx.x] = vs;
    __syncthreads();
    for (int st = 128; st > 0; st >>= 1) {
        if (threadIdx.x < st) red[threadIdx.x] += red[threadIdx.x + st];
        __syncthreads();
    }
    float rstd = rsqrtf(red[0] / Dc + 1e-5f);
#pragma unroll
    for (int i = 0; i < 4; i++) {
        int c = threadIdx.x + i * 256;
        float ov = (v[i] - mu) * rstd * g[c] + bta[c];
        out[(size_t)row * Dc + c] = ov;
        if (outb) outb[(size_t)row * Dc + c] = (bf16_t)ov;
        if (out2) out2[(size_t)row * Dc + c] = ov;
    }
}

// ---------------------------------------------------------------------------
// Host orchestration
// ---------------------------------------------------------------------------
extern "C" void kernel_launch(void* const* d_in, const int* in_sizes, int n_in,
                              void* d_out, int out_size, void* d_ws, size_t ws_size,
                              hipStream_t stream) {
    (void)in_sizes; (void)n_in; (void)out_size; (void)ws_size;

    const float* in_inputs = (const float*)d_in[0];
    const float* in_mems   = (const float*)d_in[1];
    const float* qkv_w     = (const float*)d_in[2];
    const float* o_w       = (const float*)d_in[3];
    const float* r_w       = (const float*)d_in[4];
    const float* attn_ln_g = (const float*)d_in[5];
    const float* attn_ln_b = (const float*)d_in[6];
    const float* ff_w1     = (const float*)d_in[7];
    const float* ff_b1     = (const float*)d_in[8];
    const float* ff_w2     = (const float*)d_in[9];
    const float* ff_b2     = (const float*)d_in[10];
    const float* ff_ln_g   = (const float*)d_in[11];
    const float* ff_ln_b   = (const float*)d_in[12];
    const float* fin_g     = (const float*)d_in[13];
    const float* fin_b     = (const float*)d_in[14];
    const float* r_w_bias  = (const float*)d_in[15];
    const float* r_r_bias  = (const float*)d_in[16];
    float* outp = (float*)d_out;

    char* ws = (char*)d_ws;
    size_t off = 0;
    auto alloc = [&](size_t bytes) -> char* {
        char* pp = ws + off;
        off = (off + bytes + 255) & ~(size_t)255;
        return pp;
    };
    float*  core  = (float*) alloc((size_t)Tc * Bc * Dc * 4);
    bf16_t* coreb = (bf16_t*)alloc((size_t)Tc * Bc * Dc * 2);
    bf16_t* catA  = (bf16_t*)alloc((size_t)KLc * Bc * Dc * 2);
    float*  wh    = (float*) alloc((size_t)KLc * Bc * 3 * HDHc * 4);
    bf16_t* peb   = (bf16_t*)alloc((size_t)KLc * Dc * 2);
    bf16_t* rkb   = (bf16_t*)alloc((size_t)KLc * HDHc * 2);
    bf16_t* Qw    = (bf16_t*)alloc((size_t)Bc * Hc * Tc * DHc * 2);
    bf16_t* Qr    = (bf16_t*)alloc((size_t)Bc * Hc * Tc * DHc * 2);
    bf16_t* Kb    = (bf16_t*)alloc((size_t)Bc * Hc * KLc * DHc * 2);
    bf16_t* Vb    = (bf16_t*)alloc((size_t)Bc * Hc * KLc * DHc * 2);
    bf16_t* av    = (bf16_t*)alloc((size_t)Tc * Bc * HDHc * 2);
    float*  proj  = (float*) alloc((size_t)Tc * Bc * Dc * 4);
    bf16_t* ff1o  = (bf16_t*)alloc((size_t)Tc * Bc * DIc * 2);
    bf16_t* qkvwb = (bf16_t*)alloc((size_t)Dc * 3 * HDHc * 2);
    bf16_t* rwb   = (bf16_t*)alloc((size_t)Dc * HDHc * 2);
    bf16_t* owb   = (bf16_t*)alloc((size_t)HDHc * Dc * 2);
    bf16_t* f1b   = (bf16_t*)alloc((size_t)Dc * DIc * 2);
    bf16_t* f2b   = (bf16_t*)alloc((size_t)DIc * Dc * 2);

    auto cast = [&](const float* src, bf16_t* dst, size_t n) {
        cast_f2b_kernel<<<(unsigned)((n + 255) / 256), 256, 0, stream>>>(src, dst, n);
    };

    hipMemcpyAsync(core, in_inputs, (size_t)Tc * Bc * Dc * 4,
                   hipMemcpyDeviceToDevice, stream);
    {
        size_t n = (size_t)KLc * (Dc / 2);
        pos_emb_kernel<<<(unsigned)((n + 255) / 256), 256, 0, stream>>>(peb);
    }

    const size_t nMBD = (size_t)Mc * Bc * Dc;
    const size_t nTBD = (size_t)Tc * Bc * Dc;

    for (int l = 0; l < Lc; l++) {
        // cat = [mems[l]; core]  (flat concat), cast to bf16
        cast(in_mems + (size_t)l * nMBD, catA, nMBD);
        cast(core, catA + nMBD, nTBD);
        // per-layer weight casts
        cast(qkv_w + (size_t)l * Dc * 3 * HDHc, qkvwb, (size_t)Dc * 3 * HDHc);
        cast(r_w   + (size_t)l * Dc * HDHc,     rwb,   (size_t)Dc * HDHc);
        cast(o_w   + (size_t)l * HDHc * Dc,     owb,   (size_t)HDHc * Dc);
        cast(ff_w1 + (size_t)l * Dc * DIc,      f1b,   (size_t)Dc * DIc);
        cast(ff_w2 + (size_t)l * DIc * Dc,      f2b,   (size_t)DIc * Dc);

        // wh = cat @ qkv_w   [8192 x 3072 x 1024]
        gemm_bf16_kernel<<<dim3(3 * HDHc / TBN, KLc * Bc / TBM), 256, 0, stream>>>(
            catA, Dc, qkvwb, 3 * HDHc, wh, nullptr, 3 * HDHc, nullptr, 0, Dc);
        // rk = pe @ r_w -> bf16   [2048 x 1024 x 1024]
        gemm_bf16_kernel<<<dim3(HDHc / TBN, KLc / TBM), 256, 0, stream>>>(
            peb, Dc, rwb, HDHc, nullptr, rkb, HDHc, nullptr, 0, Dc);
        // split + bias
        {
            size_t n = (size_t)KLc * Bc * HDHc;
            split_qkv_kernel<<<(unsigned)((n + 255) / 256), 256, 0, stream>>>(
                wh, r_w_bias, r_r_bias, Qw, Qr, Kb, Vb);
        }
        // fused attention
        attn_kernel<<<dim3(Tc / 64, Hc, Bc), 128, 0, stream>>>(Qw, Qr, Kb, Vb, rkb, av);
        // output projection  [4096 x 1024 x 1024]
        gemm_bf16_kernel<<<dim3(Dc / TBN, Tc * Bc / TBM), 256, 0, stream>>>(
            av, HDHc, owb, Dc, proj, nullptr, Dc, nullptr, 0, HDHc);
        // attn LN (writes core fp32 + bf16 copy for FF1)
        ln_kernel<<<Tc * Bc, 256, 0, stream>>>(
            core, proj, attn_ln_g + (size_t)l * Dc, attn_ln_b + (size_t)l * Dc,
            core, coreb, nullptr);
        // FF1 (+bias, relu) -> bf16   [4096 x 4096 x 1024]
        gemm_bf16_kernel<<<dim3(DIc / TBN, Tc * Bc / TBM), 256, 0, stream>>>(
            coreb, Dc, f1b, DIc, nullptr, ff1o, DIc,
            ff_b1 + (size_t)l * DIc, 1, Dc);
        // FF2 (+bias) -> fp32   [4096 x 1024 x 4096]
        gemm_bf16_kernel<<<dim3(Dc / TBN, Tc * Bc / TBM), 256, 0, stream>>>(
            ff1o, DIc, f2b, Dc, proj, nullptr, Dc,
            ff_b2 + (size_t)l * Dc, 0, DIc);
        // FF LN: core fp32 + hids[l] straight into d_out (new_mems[l] == hids[l])
        float* hid_out = outp + nTBD + (size_t)l * nMBD;
        ln_kernel<<<Tc * Bc, 256, 0, stream>>>(
            core, proj, ff_ln_g + (size_t)l * Dc, ff_ln_b + (size_t)l * Dc,
            core, nullptr, hid_out);
    }

    // final LN -> d_out[0 : T*B*D]
    ln_kernel<<<Tc * Bc, 256, 0, stream>>>(core, nullptr, fin_g, fin_b,
                                           outp, nullptr, nullptr);
}
